// TorchSTFT_75419625718337
// MI455X (gfx1250) — matrix operand order repaired and checked by reading, not compiled, not run on hardware
//
#include <hip/hip_runtime.h>

// ---- problem constants (match reference) ----
#define FLEN   800          // frame length == GEMM K
#define HOPSZ  200
#define NCH    802          // 2*CUT
#define BATCH  16
#define TLEN   256000
#define PADW   400
#define LPAD   256800       // padded length == OLA length
#define NF     1281         // number of frames
#define FTILES 81           // ceil(NF/16)
#define NTILES 50           // 800/16 output-column tiles
#define JROWS  400          // FLEN/2 packed pair-rows of B planes

typedef __attribute__((ext_vector_type(16))) __bf16 v16bf;
typedef __attribute__((ext_vector_type(8)))  float  v8f;

union FragBF {
    v16bf        v;
    __bf16       e[16];
    unsigned int u[8];
    uint4        q[2];
};

union PackBF {
    __bf16       e[2];
    unsigned int u;
};

__device__ __forceinline__ int reflect_idx(int s) {
    // jnp.pad 'reflect' (no edge repeat); PADW < TLEN so one bounce suffices
    s = (s < 0) ? -s : s;
    s = (s >= TLEN) ? (2 * TLEN - 2 - s) : s;
    return s;
}

// ---------------------------------------------------------------------------
// Kernel 0: zero the OLA accumulator (workspace is poisoned, not zeroed)
// ---------------------------------------------------------------------------
__global__ void zero_acc_kernel(float* __restrict__ acc, int n) {
    int i = blockIdx.x * 256 + threadIdx.x;
    if (i < n) acc[i] = 0.0f;
}

// ---------------------------------------------------------------------------
// Kernel 1: M[k][n] = sum_c fwd[c][k] * inv[c][n]  (800x800, K=802, f32),
// split into pair-packed, hi/lo-INTERLEAVED bf16 planes:
//   P[j][n] = { {bf16(M[2j][n]), bf16(M[2j+1][n])},      // .x = hi pair
//               {residual pair} }                        // .y = lo pair
// so the GEMM fetches both WMMA B dwords with one global_load_b64.
// One block per pair-row j; inv reads coalesced + L2-resident.
// ---------------------------------------------------------------------------
__global__ __launch_bounds__(256)
void build_pack_kernel(const float* __restrict__ fwd,
                       const float* __restrict__ inv,
                       uint2* __restrict__ P) {
    const int j  = blockIdx.x;                    // 0..399
    const int k0 = 2 * j;
    const int k1 = 2 * j + 1;
    for (int n = threadIdx.x; n < FLEN; n += 256) {
        float a0 = 0.0f, a1 = 0.0f;
        for (int c = 0; c < NCH; ++c) {
            const float iv = inv[c * FLEN + n];
            a0 = fmaf(fwd[c * FLEN + k0], iv, a0);
            a1 = fmaf(fwd[c * FLEN + k1], iv, a1);
        }
        const __bf16 h0 = (__bf16)a0;
        const __bf16 h1 = (__bf16)a1;
        PackBF ph, pl;
        ph.e[0] = h0;
        ph.e[1] = h1;
        pl.e[0] = (__bf16)(a0 - (float)h0);
        pl.e[1] = (__bf16)(a1 - (float)h1);
        P[j * FLEN + n] = make_uint2(ph.u, pl.u);
    }
}

// ---------------------------------------------------------------------------
// Kernel 2: contrib = frames @ M via split-bf16 WMMA (Ah*Bh + Ah*Bl + Al*Bh),
// overlap-added with f32 atomics.  grid = (4 wave-groups, 81 frame tiles, 16).
// 256 threads = 8 wave32s; each wave owns TWO 16x16 C tiles (w and w+32) so
// the LDS A slab + ds_load_b128 fragments amortize over 6 WMMAs per K-step.
// B fragments: 8 coalesced global_load_b64 per tile per K-step.
// ---------------------------------------------------------------------------
__global__ __launch_bounds__(256)
void stft_wmma_kernel(const float* __restrict__ x,
                      const uint2* __restrict__ P,
                      float* __restrict__ acc) {
    // row stride 40 (*2B = 80B) keeps every 16B fragment read aligned
    __shared__ __align__(16) __bf16 Ahi[16][40];
    __shared__ __align__(16) __bf16 Alo[16][40];

    const int b    = blockIdx.z;
    const int f0   = blockIdx.y * 16;             // first frame of tile
    const int wave = threadIdx.x >> 5;
    const int lane = threadIdx.x & 31;
    const int wid  = blockIdx.x * 8 + wave;       // 0..31
    const int nt0  = wid;                         // tile A: always valid (<50)
    const int nt1  = wid + 32;                    // tile B: valid for wid<18
    const bool two = (nt1 < NTILES);

    const int laneN = lane & 15;                  // N column / A row (M dim)
    const int hw    = lane >> 4;                  // lane half selects K group

    const float* xb = x + (size_t)b * TLEN;
    // per-lane column bases into the interleaved packed B plane
    const uint2* P0 = P + nt0 * 16 + laneN;
    const uint2* P1 = P + (two ? nt1 * 16 : 0) + laneN;

    v8f c0 = {0.f, 0.f, 0.f, 0.f, 0.f, 0.f, 0.f, 0.f};
    v8f c1 = {0.f, 0.f, 0.f, 0.f, 0.f, 0.f, 0.f, 0.f};

    for (int K = 0; K < FLEN; K += 32) {
        // ---- cooperative A staging: 16 frames x 32 samples, split bf16 ----
        {
            int id = threadIdx.x * 2;
#pragma unroll
            for (int r = 0; r < 2; ++r, ++id) {
                const int fr   = id >> 5;         // 0..15
                const int samp = id & 31;
                const int f    = f0 + fr;
                const int fc   = (f < NF) ? f : (NF - 1);   // clamp, no branch
                const int s    = reflect_idx(fc * HOPSZ + K + samp - PADW);
                float val = xb[s];
                val = (f < NF) ? val : 0.0f;                // v_cndmask
                const __bf16 h = (__bf16)val;
                Ahi[fr][samp] = h;
                Alo[fr][samp] = (__bf16)(val - (float)h);
            }
        }
        __syncthreads();

        // A fragment (16x32 bf16): VGPR0-3 = K 8*hw..+7, VGPR4-7 = K 16+8*hw..+7
        FragBF a_hi, a_lo;
        a_hi.q[0] = *(const uint4*)&Ahi[laneN][8 * hw];
        a_hi.q[1] = *(const uint4*)&Ahi[laneN][16 + 8 * hw];
        a_lo.q[0] = *(const uint4*)&Alo[laneN][8 * hw];
        a_lo.q[1] = *(const uint4*)&Alo[laneN][16 + 8 * hw];

        // pair-row base for this lane half: k = K + 2v + 8*hw (+8 for v>=4)  ->  j = K/2 + v + 4*hw (+4 for v>=4)
        const int jb = (K >> 1) + 4 * hw;

        // keep the GL2 -> WGP path warm for the next K slab of B
        if (K + 32 < FLEN) {
            __builtin_prefetch(P0 + (size_t)(jb + 16) * FLEN, 0, 3);
        }

        {   // ---- tile A (always valid) ----
            FragBF b_hi, b_lo;
#pragma unroll
            for (int v = 0; v < 8; ++v) {
                const uint2 t = P0[(size_t)(jb + v + (v >= 4 ? 4 : 0)) * FLEN];   // one b64 load
                b_hi.u[v] = t.x;
                b_lo.u[v] = t.y;
            }
            c0 = __builtin_amdgcn_wmma_f32_16x16x32_bf16(
                     false, a_hi.v, false, b_hi.v, (short)0, c0, false, false);
            c0 = __builtin_amdgcn_wmma_f32_16x16x32_bf16(
                     false, a_hi.v, false, b_lo.v, (short)0, c0, false, false);
            c0 = __builtin_amdgcn_wmma_f32_16x16x32_bf16(
                     false, a_lo.v, false, b_hi.v, (short)0, c0, false, false);
        }
        if (two) {   // ---- tile B (wave-uniform predicate) ----
            FragBF b_hi, b_lo;
#pragma unroll
            for (int v = 0; v < 8; ++v) {
                const uint2 t = P1[(size_t)(jb + v + (v >= 4 ? 4 : 0)) * FLEN];   // one b64 load
                b_hi.u[v] = t.x;
                b_lo.u[v] = t.y;
            }
            c1 = __builtin_amdgcn_wmma_f32_16x16x32_bf16(
                     false, a_hi.v, false, b_hi.v, (short)0, c1, false, false);
            c1 = __builtin_amdgcn_wmma_f32_16x16x32_bf16(
                     false, a_hi.v, false, b_lo.v, (short)0, c1, false, false);
            c1 = __builtin_amdgcn_wmma_f32_16x16x32_bf16(
                     false, a_lo.v, false, b_hi.v, (short)0, c1, false, false);
        }
        __syncthreads();
    }

    // ---- overlap-add the C tiles (layout: row M = r + 8*hw, col = laneN) ----
    float* dst0 = acc + (size_t)b * LPAD + nt0 * 16 + laneN;
    float* dst1 = acc + (size_t)b * LPAD + nt1 * 16 + laneN;
    if (f0 + 16 <= NF) {                         // block-uniform fast path
#pragma unroll
        for (int r = 0; r < 8; ++r) {
            const size_t off = (size_t)(f0 + r + 8 * hw) * HOPSZ;
            unsafeAtomicAdd(dst0 + off, c0[r]);
            if (two) unsafeAtomicAdd(dst1 + off, c1[r]);
        }
    } else {                                     // only last frame tile
#pragma unroll
        for (int r = 0; r < 8; ++r) {
            const int fr = f0 + r + 8 * hw;
            if (fr < NF) {
                const size_t off = (size_t)fr * HOPSZ;
                unsafeAtomicAdd(dst0 + off, c0[r]);
                if (two) unsafeAtomicAdd(dst1 + off, c1[r]);
            }
        }
    }
}

// ---------------------------------------------------------------------------
// Kernel 3: normalize by OLA'd window_sq, crop padding, write output
// ---------------------------------------------------------------------------
__global__ __launch_bounds__(256)
void ola_norm_kernel(const float* __restrict__ acc,
                     const float* __restrict__ wsq,
                     float* __restrict__ out) {
    const int i = blockIdx.x * 256 + threadIdx.x;   // grid sized exactly
    const int b = i / TLEN;
    const int t = i - b * TLEN;
    const int g = t + PADW;                          // position in padded domain

    int fmin = (g >= FLEN) ? ((g - FLEN) / HOPSZ + 1) : 0;
    int fmax = g / HOPSZ;
    if (fmax > NF - 1) fmax = NF - 1;

    float ws = 0.0f;
    for (int f = fmin; f <= fmax; ++f) ws += wsq[g - f * HOPSZ];
    const float d = (ws > 1e-8f) ? ws : 1.0f;

    out[i] = acc[(size_t)b * LPAD + g] / d * ((float)FLEN / (float)HOPSZ);
}

// ---------------------------------------------------------------------------
extern "C" void kernel_launch(void* const* d_in, const int* in_sizes, int n_in,
                              void* d_out, int out_size, void* d_ws, size_t ws_size,
                              hipStream_t stream) {
    const float* x   = (const float*)d_in[0];    // (16, 256000)
    const float* fwd = (const float*)d_in[1];    // (802, 800)
    const float* inv = (const float*)d_in[2];    // (802, 800)
    const float* wsq = (const float*)d_in[3];    // (800,)
    float*       out = (float*)d_out;            // (16,1,1,256000) flat

    // workspace layout: P interleaved hi/lo planes (2.56 MB) | acc (16.4 MB)
    uint2* P   = (uint2*)d_ws;
    float* acc = (float*)(P + (size_t)JROWS * FLEN);

    const int acc_n = BATCH * LPAD;                              // 4,108,800
    zero_acc_kernel<<<(acc_n + 255) / 256, 256, 0, stream>>>(acc, acc_n);

    build_pack_kernel<<<JROWS, 256, 0, stream>>>(fwd, inv, P);

    dim3 grid(4, FTILES, BATCH);                 // 32 waves cover 50 n-tiles, 2 each
    stft_wmma_kernel<<<grid, 256, 0, stream>>>(x, P, acc);

    const int total = BATCH * TLEN;              // 4,096,000 = 16000*256
    ola_norm_kernel<<<total / 256, 256, 0, stream>>>(acc, wsq, out);
}
